// ACiNet_3358664425479
// MI455X (gfx1250) — compile-verified, matrix operand order checked by reading, not run on hardware
//
#include <hip/hip_runtime.h>
#include <math.h>

// ---------------------------------------------------------------------------
// Types for CDNA5 WMMA (wave32, v_wmma_f32_16x16x32_bf16)
// ---------------------------------------------------------------------------
typedef unsigned short u16;
typedef __attribute__((ext_vector_type(16))) __bf16        v16bf;
typedef __attribute__((ext_vector_type(8)))  float         v8f;
typedef __attribute__((ext_vector_type(4)))  unsigned int  v4u;

union BFrag { v16bf v; v4u q[2]; };

__device__ __forceinline__ u16 f2bf(float f) {
  unsigned int u = __float_as_uint(f);
  unsigned int r = u + 0x7FFFu + ((u >> 16) & 1u);   // round-to-nearest-even
  return (u16)(r >> 16);
}

__device__ __forceinline__ v8f vzero8() {
  v8f z;
  for (int i = 0; i < 8; ++i) z[i] = 0.f;
  return z;
}

#define CDIV(a, b) (((a) + (b) - 1) / (b))

// Model constants
#define BB   32
#define BCAL 16
#define LL   2048
#define SS   512
#define DD   81
#define DP   96      // D padded to K-multiple of 32
#define HH   9
#define DK   9
#define FF   256
#define NR   (BB * SS)      // 16384 rows of (b,s)
#define NRC  ((BB + BCAL) * SS)

// ---------------------------------------------------------------------------
// 1) global min / max of x[:, 20:2028]  ->  stats[0]=xm, stats[1]=1/xs
// ---------------------------------------------------------------------------
__global__ __launch_bounds__(256) void k_minmax(const float* __restrict__ x,
                                                float* __restrict__ stats) {
  __shared__ float smn[256], smx[256];
  int tid = threadIdx.x;
  float mn = 1e30f, mx = -1e30f;
  const int n = BB * 2008;
  for (int i = tid; i < n; i += 256) {
    int b = i / 2008, j = i % 2008;
    float v = x[b * LL + 20 + j];
    mn = fminf(mn, v);
    mx = fmaxf(mx, v);
  }
  smn[tid] = mn; smx[tid] = mx;
  __syncthreads();
  for (int o = 128; o > 0; o >>= 1) {
    if (tid < o) { smn[tid] = fminf(smn[tid], smn[tid + o]);
                   smx[tid] = fmaxf(smx[tid], smx[tid + o]); }
    __syncthreads();
  }
  if (tid == 0) {
    stats[0] = smn[0];
    stats[1] = 1.f / fabsf(smx[0] - smn[0]);
  }
}

// ---------------------------------------------------------------------------
// 2) tokenize + positional encoding -> tx f32 (48,512,81) and bf16 (48,512,96)
// ---------------------------------------------------------------------------
__global__ __launch_bounds__(256) void k_tokenize(const float* __restrict__ x,
                                                  const float* __restrict__ cali,
                                                  const float* __restrict__ stats,
                                                  float* __restrict__ txf,
                                                  u16* __restrict__ txb) {
  size_t i = (size_t)blockIdx.x * 256 + threadIdx.x;
  const size_t total = (size_t)(BB + BCAL) * SS * DP;
  if (i >= total) return;
  int d = (int)(i % DP);
  size_t rest = i / DP;
  int s = (int)(rest % SS);
  int src = (int)(rest / SS);
  if (d >= DD) { txb[i] = 0; return; }
  int pos = s * 4 + d - 40;
  float v = 0.f;
  if (pos >= 0 && pos < LL) {
    float raw = (src < BB) ? x[(size_t)src * LL + pos]
                           : cali[(size_t)(src - BB) * LL + pos];
    v = (raw - stats[0]) * stats[1];
  }
  int i2 = (d >> 1) * 2;
  float div = expf(-logf(10000.f) * (float)i2 / (float)DD);
  float ang = (float)s * div;
  v += (d & 1) ? cosf(ang) : sinf(ang);
  txf[((size_t)src * SS + s) * DD + d] = v;
  txb[i] = f2bf(v);
}

// ---------------------------------------------------------------------------
// 3) bs = basel[:, ::4]
// ---------------------------------------------------------------------------
__global__ __launch_bounds__(256) void k_bs(const float* __restrict__ basel,
                                            float* __restrict__ bs) {
  int i = blockIdx.x * 256 + threadIdx.x;
  if (i >= BB * SS) return;
  int b = i / SS, s = i % SS;
  bs[i] = basel[(size_t)b * LL + s * 4];
}

// ---------------------------------------------------------------------------
// 4) pack weight -> bf16 transposed (Np x Kp), zero-padded.
//    mode 0: src is (K x N) row-major f32
//    mode 1: src is (H, K, dk) "hed" projection, N = H*dk
// ---------------------------------------------------------------------------
__global__ __launch_bounds__(256) void k_packw(const float* __restrict__ src,
                                               u16* __restrict__ dst,
                                               int K, int N, int Kp, int Np,
                                               int mode) {
  size_t i = (size_t)blockIdx.x * 256 + threadIdx.x;
  size_t total = (size_t)Np * Kp;
  if (i >= total) return;
  int n = (int)(i / Kp), k = (int)(i % Kp);
  float v = 0.f;
  if (n < N && k < K) {
    v = mode ? src[(((size_t)(n / DK)) * K + k) * DK + (n % DK)]
             : src[(size_t)k * N + n];
  }
  dst[i] = f2bf(v);
}

// ---------------------------------------------------------------------------
// 5) pack activation f32 (M x K) -> bf16 (M x Kp), zero-padded K
// ---------------------------------------------------------------------------
__global__ __launch_bounds__(256) void k_packa(const float* __restrict__ src,
                                               u16* __restrict__ dst,
                                               int K, int Kp, size_t total) {
  size_t i = (size_t)blockIdx.x * 256 + threadIdx.x;
  if (i >= total) return;
  size_t m = i / Kp;
  int k = (int)(i % Kp);
  dst[i] = (k < K) ? f2bf(src[m * K + k]) : (u16)0;
}

// ---------------------------------------------------------------------------
// 6) register-blocked WMMA GEMM:
//    C(MxN,f32) = alpha * A(MxKp,bf16) * Bt(NpxKp,bf16)^T + bias[n], opt relu.
//    One wave computes a 16 x (16*NT) strip: the A fragment is loaded once per
//    K-step and reused across NT independent accumulators (4 back-to-back
//    v_wmma issues deepen the XDL pipeline and cut A-operand cache traffic).
//    Requires M%16==0, Kp%32==0, grid.x = Np/(16*NT).
// ---------------------------------------------------------------------------
template <int NT>
__global__ __launch_bounds__(32)
void k_gemmT(const u16* __restrict__ A, const u16* __restrict__ Bt,
             float* __restrict__ C, int M, int N, int Kp,
             const float* __restrict__ bias, float alpha, int relu) {
  int lane = threadIdx.x;
  int half = lane >> 4, idx = lane & 15;
  const u16* ap  = A  + ((size_t)blockIdx.y * 16 + idx) * Kp + half * 8;
  const u16* bp0 = Bt + (((size_t)blockIdx.x * NT) * 16 + idx) * Kp + half * 8;
  v8f acc[NT];
#pragma unroll
  for (int t = 0; t < NT; ++t) acc[t] = vzero8();
  for (int k0 = 0; k0 < Kp; k0 += 32) {
    BFrag fa;
    fa.q[0] = *(const v4u*)(ap + k0);
    fa.q[1] = *(const v4u*)(ap + k0 + 16);
#pragma unroll
    for (int t = 0; t < NT; ++t) {
      BFrag fb;
      const u16* bp = bp0 + (size_t)t * 16 * Kp;
      fb.q[0] = *(const v4u*)(bp + k0);
      fb.q[1] = *(const v4u*)(bp + k0 + 16);
      acc[t] = __builtin_amdgcn_wmma_f32_16x16x32_bf16(false, fa.v, false, fb.v,
                                                       (short)0, acc[t], false, false);
    }
  }
#pragma unroll
  for (int t = 0; t < NT; ++t) {
    int n = (blockIdx.x * NT + t) * 16 + idx;
    if (n < N) {
      float bv = bias ? bias[n] : 0.f;
      for (int r = 0; r < 8; ++r) {
        int m = blockIdx.y * 16 + half * 8 + r;
        float v = alpha * acc[t][r] + bv;
        if (relu) v = fmaxf(v, 0.f);
        C[(size_t)m * N + n] = v;
      }
    }
  }
}

// ---------------------------------------------------------------------------
// 7) attention-1 (tiny masked cross-attention to self + 16 cali keys). VALU.
//    grid = S blocks, 288 threads (h = t>>5, b = t&31)
// ---------------------------------------------------------------------------
__global__ __launch_bounds__(288) void k_attn1(const float* __restrict__ q2,
                                               const float* __restrict__ k2,
                                               const float* __restrict__ v2,
                                               float* __restrict__ h2pre) {
  int s = blockIdx.x;
  int t = threadIdx.x;
  int h = t >> 5, b = t & 31;
  size_t qoff = ((size_t)b * SS + s) * DD + h * DK;
  float q[DK];
  for (int d = 0; d < DK; ++d) q[d] = q2[qoff + d];
  float logit[17];
  float mx = -1e30f;
  for (int ki = 0; ki < 17; ++ki) {
    int src = (ki == 0) ? b : (BB + ki - 1);
    const float* kr = k2 + ((size_t)src * SS + s) * DD + h * DK;
    float dot = 0.f;
    for (int j = 0; j < DK; ++j) dot += q[j] * kr[j];
    logit[ki] = dot * (1.f / 3.f);
    mx = fmaxf(mx, logit[ki]);
  }
  float sum = 0.f;
  for (int ki = 0; ki < 17; ++ki) { logit[ki] = __expf(logit[ki] - mx); sum += logit[ki]; }
  float inv = 1.f / sum;
  float out[DK];
  for (int j = 0; j < DK; ++j) out[j] = 0.f;
  for (int ki = 0; ki < 17; ++ki) {
    int src = (ki == 0) ? b : (BB + ki - 1);
    const float* vr = v2 + ((size_t)src * SS + s) * DD + h * DK;
    float p = logit[ki] * inv;
    for (int j = 0; j < DK; ++j) out[j] += p * vr[j];
  }
  for (int j = 0; j < DK; ++j) h2pre[qoff + j] = out[j];
}

// ---------------------------------------------------------------------------
// 8) elementwise: c = a + b
// ---------------------------------------------------------------------------
__global__ __launch_bounds__(256) void k_add(const float* __restrict__ a,
                                             const float* __restrict__ b,
                                             float* __restrict__ c, size_t n) {
  size_t i = (size_t)blockIdx.x * 256 + threadIdx.x;
  if (i < n) c[i] = a[i] + b[i];
}

// 8b) att + residual mix: out = (1-hw)*h1 + hw*h2 + tx
__global__ __launch_bounds__(256) void k_mix(const float* __restrict__ h1,
                                             const float* __restrict__ h2,
                                             const float* __restrict__ tx,
                                             const float* __restrict__ hw,
                                             float* __restrict__ out, size_t n) {
  size_t i = (size_t)blockIdx.x * 256 + threadIdx.x;
  if (i >= n) return;
  float w = hw[0];
  out[i] = (1.f - w) * h1[i] + w * h2[i] + tx[i];
}

// ---------------------------------------------------------------------------
// 9) head pack for attention-2: qh/kh (B,H,512,32) bf16 (K=dk padded to 32)
// ---------------------------------------------------------------------------
__global__ __launch_bounds__(256) void k_packheads(const float* __restrict__ qs,
                                                   const float* __restrict__ ks,
                                                   u16* __restrict__ qh,
                                                   u16* __restrict__ kh) {
  size_t i = (size_t)blockIdx.x * 256 + threadIdx.x;
  const size_t total = (size_t)BB * HH * SS * 32;
  if (i >= total) return;
  int k = (int)(i & 31);
  size_t rest = i >> 5;
  int s = (int)(rest % SS); rest /= SS;
  int h = (int)(rest % HH);
  int b = (int)(rest / HH);
  if (k < DK) {
    size_t src = ((size_t)b * SS + s) * DD + h * DK + k;
    qh[i] = f2bf(qs[src]);
    kh[i] = f2bf(ks[src]);
  } else { qh[i] = 0; kh[i] = 0; }
}

// V^T pack (B,H,16,512): vt[bh][d][s]  (Vs == Qs in the reference)
__global__ __launch_bounds__(256) void k_packvt(const float* __restrict__ qs,
                                                u16* __restrict__ vt) {
  size_t i = (size_t)blockIdx.x * 256 + threadIdx.x;
  const size_t total = (size_t)BB * HH * 16 * SS;
  if (i >= total) return;
  int s = (int)(i % SS);
  size_t rest = i / SS;
  int d = (int)(rest % 16); rest /= 16;
  int h = (int)(rest % HH);
  int b = (int)(rest / HH);
  vt[i] = (d < DK) ? f2bf(qs[((size_t)b * SS + s) * DD + h * DK + d]) : (u16)0;
}

// ---------------------------------------------------------------------------
// 10) fused attention-2: one wave per (b, h, 16-row s-tile).
//     scores (16x512) live in LDS, softmax in-place, then prob @ V via WMMA.
// ---------------------------------------------------------------------------
__global__ __launch_bounds__(32) void k_attn2(const u16* __restrict__ qh,
                                              const u16* __restrict__ kh,
                                              const u16* __restrict__ vt,
                                              const float* __restrict__ biasM,
                                              const float* __restrict__ corr_weight,
                                              float* __restrict__ h1pre) {
  __shared__ float Ssc[16 * SS];   // 32 KB
  __shared__ u16  Pp[16 * SS];     // 16 KB
  int st = blockIdx.x;
  int h  = blockIdx.y;
  int b  = blockIdx.z;
  int lane = threadIdx.x;
  int half = lane >> 4, idx = lane & 15;
  size_t bh = (size_t)b * HH + h;
  const u16* qb = qh + bh * SS * 32;
  const u16* kb = kh + bh * SS * 32;
  const u16* vb = vt + bh * 16 * SS;
  float cw = corr_weight[0];
  float qscale = (1.f - cw) * (1.f / 3.f);

  BFrag fa;
  {
    const u16* pa = qb + ((size_t)st * 16 + idx) * 32 + half * 8;
    fa.q[0] = *(const v4u*)pa;
    fa.q[1] = *(const v4u*)(pa + 16);
  }
  for (int tt = 0; tt < SS / 16; ++tt) {
    BFrag fb;
    const u16* pb = kb + ((size_t)tt * 16 + idx) * 32 + half * 8;
    fb.q[0] = *(const v4u*)pb;
    fb.q[1] = *(const v4u*)(pb + 16);
    v8f acc = vzero8();
    acc = __builtin_amdgcn_wmma_f32_16x16x32_bf16(false, fa.v, false, fb.v,
                                                  (short)0, acc, false, false);
    int key = tt * 16 + idx;
    for (int r = 0; r < 8; ++r) {
      int srow = half * 8 + r;
      Ssc[srow * SS + key] =
          qscale * acc[r] + cw * biasM[(size_t)(st * 16 + srow) * SS + key];
    }
  }
  __syncthreads();
  if (lane < 16) {
    float* row = Ssc + lane * SS;
    float mx = -1e30f;
    for (int t2 = 0; t2 < SS; ++t2) mx = fmaxf(mx, row[t2]);
    float sum = 0.f;
    for (int t2 = 0; t2 < SS; ++t2) { float e = __expf(row[t2] - mx); row[t2] = e; sum += e; }
    float inv = 1.f / sum;
    u16* prow = Pp + lane * SS;
    for (int t2 = 0; t2 < SS; ++t2) prow[t2] = f2bf(row[t2] * inv);
  }
  __syncthreads();
  v8f c = vzero8();
  for (int kk = 0; kk < SS / 32; ++kk) {
    BFrag pa, pv;
    const u16* pp = Pp + (size_t)idx * SS + kk * 32 + half * 8;
    pa.q[0] = *(const v4u*)pp;
    pa.q[1] = *(const v4u*)(pp + 16);
    const u16* vv = vb + (size_t)idx * SS + kk * 32 + half * 8;
    pv.q[0] = *(const v4u*)vv;
    pv.q[1] = *(const v4u*)(vv + 16);
    c = __builtin_amdgcn_wmma_f32_16x16x32_bf16(false, pa.v, false, pv.v,
                                                (short)0, c, false, false);
  }
  if (idx < DK) {
    for (int r = 0; r < 8; ++r) {
      int s = st * 16 + half * 8 + r;
      h1pre[((size_t)b * SS + s) * DD + h * DK + idx] = c[r];
    }
  }
}

// ---------------------------------------------------------------------------
// 11) batchnorm over rows: per-channel mean & rstd
// ---------------------------------------------------------------------------
__global__ __launch_bounds__(256) void k_bnstats(const float* __restrict__ x,
                                                 float* __restrict__ mean,
                                                 float* __restrict__ rstd,
                                                 int C, int Nrow) {
  __shared__ float s1[256], s2[256];
  int c = blockIdx.x, tid = threadIdx.x;
  float a = 0.f, b = 0.f;
  for (int r = tid; r < Nrow; r += 256) {
    float v = x[(size_t)r * C + c];
    a += v; b += v * v;
  }
  s1[tid] = a; s2[tid] = b;
  __syncthreads();
  for (int o = 128; o > 0; o >>= 1) {
    if (tid < o) { s1[tid] += s1[tid + o]; s2[tid] += s2[tid + o]; }
    __syncthreads();
  }
  if (tid == 0) {
    float m = s1[0] / (float)Nrow;
    float v = s2[0] / (float)Nrow - m * m;
    mean[c] = m;
    rstd[c] = rsqrtf(v + 1e-5f);
  }
}

__global__ __launch_bounds__(256) void k_bnapply(const float* __restrict__ x,
                                                 const float* __restrict__ mean,
                                                 const float* __restrict__ rstd,
                                                 const float* __restrict__ g,
                                                 const float* __restrict__ bta,
                                                 float* __restrict__ outf,
                                                 u16* __restrict__ outbf,
                                                 int C, int Cp, int Nrow) {
  size_t i = (size_t)blockIdx.x * 256 + threadIdx.x;
  size_t total = (size_t)Nrow * Cp;
  if (i >= total) return;
  size_t r = i / Cp;
  int k = (int)(i % Cp);
  if (k < C) {
    float v = (x[r * C + k] - mean[k]) * rstd[k] * g[k] + bta[k];
    if (outf)  outf[r * C + k] = v;
    if (outbf) outbf[r * Cp + k] = f2bf(v);
  } else if (outbf) {
    outbf[r * Cp + k] = 0;
  }
}

// ---------------------------------------------------------------------------
// 12) token aggregation: a[r] = sum_d tv[r,d]*softmax(btm[r,:])[d];
//     fea = alpha1*a + alpha2*bs
// ---------------------------------------------------------------------------
__global__ __launch_bounds__(256) void k_tokenagg(const float* __restrict__ tv,
                                                  const float* __restrict__ btm,
                                                  const float* __restrict__ bs,
                                                  const float* __restrict__ a1,
                                                  const float* __restrict__ a2,
                                                  float* __restrict__ fea) {
  int r = blockIdx.x * 256 + threadIdx.x;
  if (r >= NR) return;
  const float* lb = btm + (size_t)r * DD;
  float mx = -1e30f;
  for (int d = 0; d < DD; ++d) mx = fmaxf(mx, lb[d]);
  float sum = 0.f;
  for (int d = 0; d < DD; ++d) sum += __expf(lb[d] - mx);
  float inv = 1.f / sum;
  float acc = 0.f;
  const float* lt = tv + (size_t)r * DD;
  for (int d = 0; d < DD; ++d) acc += lt[d] * (__expf(lb[d] - mx) * inv);
  fea[r] = a1[0] * acc + a2[0] * bs[r];
}

// ---------------------------------------------------------------------------
// 13) small dense layer: out(BxN) = act(in(BxK) @ W(KxN) + bias)
//     act: 0 none, 1 relu, 2 tanh
// ---------------------------------------------------------------------------
__global__ __launch_bounds__(256) void k_fc(const float* __restrict__ in,
                                            const float* __restrict__ W,
                                            const float* __restrict__ bias,
                                            float* __restrict__ out,
                                            int Brow, int K, int N, int act) {
  int i = blockIdx.x * 256 + threadIdx.x;
  if (i >= Brow * N) return;
  int b = i / N, n = i % N;
  float acc = bias ? bias[n] : 0.f;
  for (int k = 0; k < K; ++k) acc += in[(size_t)b * K + k] * W[(size_t)k * N + n];
  if (act == 1) acc = fmaxf(acc, 0.f);
  if (act == 2) acc = tanhf(acc);
  out[i] = acc;
}

// batchnorm over axis 0 for small (R x C)
__global__ __launch_bounds__(256) void k_bnrow(const float* __restrict__ x,
                                               const float* __restrict__ g,
                                               const float* __restrict__ bta,
                                               float* __restrict__ y,
                                               int C, int R) {
  int c = blockIdx.x * 256 + threadIdx.x;
  if (c >= C) return;
  float s = 0.f, q = 0.f;
  for (int r = 0; r < R; ++r) { float v = x[(size_t)r * C + c]; s += v; q += v * v; }
  float m = s / (float)R;
  float var = q / (float)R - m * m;
  float rs = rsqrtf(var + 1e-5f);
  for (int r = 0; r < R; ++r)
    y[(size_t)r * C + c] = (x[(size_t)r * C + c] - m) * rs * g[c] + bta[c];
}

// ---------------------------------------------------------------------------
// host-side orchestration
// ---------------------------------------------------------------------------
extern "C" void kernel_launch(void* const* d_in, const int* in_sizes, int n_in,
                              void* d_out, int out_size, void* d_ws, size_t ws_size,
                              hipStream_t stream) {
  (void)in_sizes; (void)n_in; (void)out_size; (void)ws_size;

  const float* x        = (const float*)d_in[0];
  const float* basel    = (const float*)d_in[1];
  const float* cali     = (const float*)d_in[2];
  const float* Wq       = (const float*)d_in[3];
  const float* Wk       = (const float*)d_in[4];
  const float* Wq2      = (const float*)d_in[5];
  const float* Wk2      = (const float*)d_in[6];
  const float* Wv2      = (const float*)d_in[7];
  const float* Cv       = (const float*)d_in[8];
  const float* Wo1      = (const float*)d_in[9];
  const float* Wo2      = (const float*)d_in[10];
  const float* corr_w   = (const float*)d_in[11];
  const float* h_w      = (const float*)d_in[12];
  const float* corr_map = (const float*)d_in[13];
  const float* g1 = (const float*)d_in[14]; const float* b1 = (const float*)d_in[15];
  const float* ff1_w = (const float*)d_in[16]; const float* ff1_b = (const float*)d_in[17];
  const float* g2 = (const float*)d_in[18]; const float* b2 = (const float*)d_in[19];
  const float* ff2_w = (const float*)d_in[20]; const float* ff2_b = (const float*)d_in[21];
  const float* g3 = (const float*)d_in[22]; const float* b3 = (const float*)d_in[23];
  const float* tv_w = (const float*)d_in[24]; const float* tv_b = (const float*)d_in[25];
  const float* beta1 = (const float*)d_in[26];
  const float* al1 = (const float*)d_in[27]; const float* al2 = (const float*)d_in[28];
  const float* fc1_w = (const float*)d_in[29]; const float* fc1_b = (const float*)d_in[30];
  const float* bnf1_g = (const float*)d_in[31]; const float* bnf1_b = (const float*)d_in[32];
  const float* fc2_w = (const float*)d_in[33]; const float* fc2_b = (const float*)d_in[34];
  const float* bnf2_g = (const float*)d_in[35]; const float* bnf2_b = (const float*)d_in[36];
  const float* fc3_w = (const float*)d_in[37]; const float* fc3_b = (const float*)d_in[38];

  // ---- bump allocator over d_ws -------------------------------------------
  char* base = (char*)d_ws;
  size_t off = 0;
  auto alloc = [&](size_t bytes) -> char* {
    char* p = base + off;
    off += (bytes + 255) & ~(size_t)255;
    return p;
  };

  float* stats = (float*)alloc(2 * 4);
  float* txf   = (float*)alloc((size_t)NRC * DD * 4);
  u16*   txb   = (u16*)  alloc((size_t)NRC * DP * 2);
  float* bs    = (float*)alloc((size_t)NR * 4);

  u16* Wq2t = (u16*)alloc(DP * DP * 2);
  u16* Wk2t = (u16*)alloc(DP * DP * 2);
  u16* Wv2t = (u16*)alloc(DP * DP * 2);
  u16* Wqt  = (u16*)alloc(DP * DP * 2);
  u16* Wkt  = (u16*)alloc(DP * DP * 2);
  u16* Wo1t = (u16*)alloc(DP * DP * 2);
  u16* Wo2t = (u16*)alloc(DP * DP * 2);
  u16* ff1t = (u16*)alloc(FF * DP * 2);
  u16* ff2t = (u16*)alloc(DP * FF * 2);
  u16* tvt  = (u16*)alloc(DP * DP * 2);
  u16* betat= (u16*)alloc(DP * DP * 2);
  u16* corrb= (u16*)alloc((size_t)SS * SS * 2);
  u16* cvt  = (u16*)alloc((size_t)SS * SS * 2);

  float* q2 = (float*)alloc((size_t)NR  * DD * 4);   // reused as f1 later
  float* k2 = (float*)alloc((size_t)NRC * DD * 4);
  float* v2 = (float*)alloc((size_t)NRC * DD * 4);
  float* h2pre  = (float*)alloc((size_t)NR * DD * 4); // reused as f1_bf later
  u16*   h2preb = (u16*)  alloc((size_t)NR * DP * 2);
  float* h2     = (float*)alloc((size_t)NR * DD * 4);
  float* xh     = (float*)alloc((size_t)NR * DD * 4);
  u16*   xhb    = (u16*)  alloc((size_t)NR * DP * 2);
  float* qs     = (float*)alloc((size_t)NR * DD * 4);
  float* ks     = (float*)alloc((size_t)NR * DD * 4);
  u16*   qh     = (u16*)  alloc((size_t)BB * HH * SS * 32 * 2);  // reused as tv later
  u16*   kh     = (u16*)  alloc((size_t)BB * HH * SS * 32 * 2);  // reused as btm later
  u16*   vtb    = (u16*)  alloc((size_t)BB * HH * 16 * SS * 2);
  float* biasM  = (float*)alloc((size_t)SS * SS * 4);
  float* h1pre  = (float*)alloc((size_t)NR * DD * 4);
  u16*   h1preb = (u16*)  alloc((size_t)NR * DP * 2);
  float* h1     = (float*)alloc((size_t)NR * DD * 4);
  float* ma_in  = (float*)alloc((size_t)NR * DD * 4);  // reused as res_in
  float* ma     = (float*)alloc((size_t)NR * DD * 4);
  u16*   mab    = (u16*)  alloc((size_t)NR * DP * 2);
  float* f2buf  = (float*)alloc((size_t)NR * DD * 4);
  u16*   resb   = (u16*)  alloc((size_t)NR * DP * 2);
  float* fea    = (float*)alloc((size_t)NR * 4);
  float* fcb1   = (float*)alloc((size_t)BB * 512 * 4);
  float* hn1    = (float*)alloc((size_t)BB * 512 * 4);
  float* fcb2   = (float*)alloc((size_t)BB * 256 * 4);
  float* hn2    = (float*)alloc((size_t)BB * 256 * 4);
  float* bmean  = (float*)alloc(256 * 4);
  float* brstd  = (float*)alloc(256 * 4);

  // explicit reuse of dead regions (phases are strictly ordered on `stream`)
  float* f1    = q2;                 // 16 MB fits in q2+k2+v2 (21.2 MB), dead after attn1
  u16*   f1b   = (u16*)h2pre;        // 8.4 MB fits in h2pre+h2preb (8.45 MB)
  float* tvbuf = (float*)qh;         // qh/kh dead after attn2
  float* btm   = (float*)kh;
  float* resin = ma_in;              // ma_in dead once ma exists

  // ---- helper lambdas ------------------------------------------------------
  auto gemm = [&](const u16* A, const u16* Bt, float* C, int M, int N, int Np,
                  int Kp, const float* bias, float alpha, int relu) {
    if (Np % 64 == 0) {
      dim3 g(Np / 64, M / 16);
      k_gemmT<4><<<g, dim3(32), 0, stream>>>(A, Bt, C, M, N, Kp, bias, alpha, relu);
    } else {
      dim3 g(Np / 32, M / 16);
      k_gemmT<2><<<g, dim3(32), 0, stream>>>(A, Bt, C, M, N, Kp, bias, alpha, relu);
    }
  };
  auto packw = [&](const float* src, u16* dst, int K, int N, int Kp, int Np, int mode) {
    size_t total = (size_t)Np * Kp;
    k_packw<<<CDIV(total, (size_t)256), 256, 0, stream>>>(src, dst, K, N, Kp, Np, mode);
  };
  auto packa = [&](const float* src, u16* dst, size_t M, int K, int Kp) {
    size_t total = M * Kp;
    k_packa<<<CDIV(total, (size_t)256), 256, 0, stream>>>(src, dst, K, Kp, total);
  };

  // ---- stage 0: stats, tokenize, baseline ---------------------------------
  k_minmax<<<1, 256, 0, stream>>>(x, stats);
  k_tokenize<<<CDIV((size_t)NRC * DP, (size_t)256), 256, 0, stream>>>(x, cali, stats, txf, txb);
  k_bs<<<CDIV(NR, 256), 256, 0, stream>>>(basel, bs);

  // ---- stage 1: weight packing (bf16, transposed, K-padded) ---------------
  packw(Wq2, Wq2t, DD, DD, DP, DP, 1);
  packw(Wk2, Wk2t, DD, DD, DP, DP, 1);
  packw(Wv2, Wv2t, DD, DD, DP, DP, 1);
  packw(Wq,  Wqt,  DD, DD, DP, DP, 1);
  packw(Wk,  Wkt,  DD, DD, DP, DP, 1);
  packw(Wo1, Wo1t, DD, DD, DP, DP, 0);
  packw(Wo2, Wo2t, DD, DD, DP, DP, 0);
  packw(ff1_w, ff1t, DD, FF, DP, FF, 0);
  packw(ff2_w, ff2t, FF, DD, FF, DP, 0);
  packw(tv_w,  tvt,  DD, DD, DP, DP, 0);
  packw(beta1, betat,DD, DD, DP, DP, 0);
  packw(Cv, cvt, SS, SS, SS, SS, 0);
  packa(corr_map, corrb, SS, SS, SS);

  // ---- stage 2: projections for attention-1 (WMMA GEMMs) ------------------
  gemm(txb, Wq2t, q2, NR,  DD, DP, DP, nullptr, 1.f, 0);
  gemm(txb, Wk2t, k2, NRC, DD, DP, DP, nullptr, 1.f, 0);
  gemm(txb, Wv2t, v2, NRC, DD, DP, DP, nullptr, 1.f, 0);

  // ---- stage 3: attention-1 + Wo2 -----------------------------------------
  k_attn1<<<SS, 288, 0, stream>>>(q2, k2, v2, h2pre);
  packa(h2pre, h2preb, NR, DD, DP);
  gemm(h2preb, Wo2t, h2, NR, DD, DP, DP, nullptr, 1.f, 0);

  // ---- stage 4: xh = tx + h2; projections for attention-2 -----------------
  k_add<<<CDIV((size_t)NR * DD, (size_t)256), 256, 0, stream>>>(txf, h2, xh, (size_t)NR * DD);
  packa(xh, xhb, NR, DD, DP);
  gemm(xhb, Wqt, qs, NR, DD, DP, DP, nullptr, 1.f, 0);   // Qs (and Vs: same weight)
  gemm(xhb, Wkt, ks, NR, DD, DP, DP, nullptr, 1.f, 0);   // Ks

  // ---- stage 5: correlation bias = corr_map @ Cv / sqrt(S) ----------------
  gemm(corrb, cvt, biasM, SS, SS, SS, SS, nullptr, 0.044194173824159216f, 0);

  // ---- stage 6: fused attention-2 (WMMA QK^T -> LDS softmax -> WMMA AV) ---
  {
    size_t th = (size_t)BB * HH * SS * 32;
    k_packheads<<<CDIV(th, (size_t)256), 256, 0, stream>>>(qs, ks, qh, kh);
    size_t tvn = (size_t)BB * HH * 16 * SS;
    k_packvt<<<CDIV(tvn, (size_t)256), 256, 0, stream>>>(qs, vtb);
    k_attn2<<<dim3(SS / 16, HH, BB), 32, 0, stream>>>(qh, kh, vtb, biasM, corr_w, h1pre);
  }
  packa(h1pre, h1preb, NR, DD, DP);
  gemm(h1preb, Wo1t, h1, NR, DD, DP, DP, nullptr, 1.f, 0);

  // ---- stage 7: mix + BN1 --------------------------------------------------
  k_mix<<<CDIV((size_t)NR * DD, (size_t)256), 256, 0, stream>>>(h1, h2, txf, h_w, ma_in,
                                                                (size_t)NR * DD);
  k_bnstats<<<DD, 256, 0, stream>>>(ma_in, bmean, brstd, DD, NR);
  k_bnapply<<<CDIV((size_t)NR * DP, (size_t)256), 256, 0, stream>>>(
      ma_in, bmean, brstd, g1, b1, ma, mab, DD, DP, NR);

  // ---- stage 8: feed-forward (WMMA + relu epilogue) -----------------------
  gemm(mab, ff1t, f1, NR, FF, FF, DP, ff1_b, 1.f, 1);    // relu(ma @ ff1 + b)
  k_bnstats<<<FF, 256, 0, stream>>>(f1, bmean, brstd, FF, NR);
  k_bnapply<<<CDIV((size_t)NR * FF, (size_t)256), 256, 0, stream>>>(
      f1, bmean, brstd, g2, b2, nullptr, f1b, FF, FF, NR);
  gemm(f1b, ff2t, f2buf, NR, DD, DP, FF, ff2_b, 1.f, 1); // relu(f @ ff2 + b)

  // ---- stage 9: res = BN(ma + f) ------------------------------------------
  k_add<<<CDIV((size_t)NR * DD, (size_t)256), 256, 0, stream>>>(ma, f2buf, resin,
                                                                (size_t)NR * DD);
  k_bnstats<<<DD, 256, 0, stream>>>(resin, bmean, brstd, DD, NR);
  k_bnapply<<<CDIV((size_t)NR * DP, (size_t)256), 256, 0, stream>>>(
      resin, bmean, brstd, g3, b3, nullptr, resb, DD, DP, NR);

  // ---- stage 10: token value / gate, aggregate ----------------------------
  gemm(resb, tvt,  tvbuf, NR, DD, DP, DP, tv_b,   1.f, 0);
  gemm(resb, betat, btm,  NR, DD, DP, DP, nullptr, 1.f, 0);
  k_tokenagg<<<CDIV(NR, 256), 256, 0, stream>>>(tvbuf, btm, bs, al1, al2, fea);

  // ---- stage 11: head MLP --------------------------------------------------
  k_fc<<<CDIV(BB * 512, 256), 256, 0, stream>>>(fea, fc1_w, fc1_b, fcb1, BB, SS, 512, 1);
  k_bnrow<<<CDIV(512, 256), 256, 0, stream>>>(fcb1, bnf1_g, bnf1_b, hn1, 512, BB);
  k_fc<<<CDIV(BB * 256, 256), 256, 0, stream>>>(hn1, fc2_w, fc2_b, fcb2, BB, 512, 256, 1);
  k_bnrow<<<CDIV(256, 256), 256, 0, stream>>>(fcb2, bnf2_g, bnf2_b, hn2, 256, BB);
  k_fc<<<1, 256, 0, stream>>>(hn2, fc3_w, fc3_b, (float*)d_out, BB, 256, 4, 2);
}
